// Transformer_72550587564543
// MI455X (gfx1250) — matrix operand order repaired and checked by reading, not compiled, not run on hardware
//
#include <hip/hip_runtime.h>
#include <hip/hip_bf16.h>

typedef __bf16 bf16;
typedef __attribute__((ext_vector_type(16))) __bf16 v16bf;
typedef __attribute__((ext_vector_type(8)))  float  v8f;

#define LDA_LDS 40    // 32  + 8 bf16 pad (80B rows, 16B aligned)
#define LDB_LDS 136   // 128 + 8 bf16 pad (272B rows, 16B aligned)

// ---------------------------------------------------------------------------
// Generic batched GEMM: C[M,N] = alpha * A_bf16[M,K] x B[K,N] + bias, opt ReLU.
// B source is fp32 (converted to bf16 while staging to LDS) or bf16.
// Outputs: optional fp32 C and/or bf16 Cb (bf16 mirror feeds next GEMM).
// Block: 256 thr = 8 waves in 4(M) x 2(N); tile 128(M) x 128(N), K step 32.
// Each wave: 32 rows x 64 cols = 2 A-frags x 4 B-frags
//   -> 8 v_wmma_f32_16x16x32_bf16 per K step (4 ds_load_b128 for A + 64 16-bit B reads per 8 WMMA).
// ---------------------------------------------------------------------------
__global__ __launch_bounds__(256)
void gemm_bf16_wmma(const bf16* __restrict__ A, long long sA, int lda,
                    const float* __restrict__ Bf, const bf16* __restrict__ Bh,
                    long long sB, int ldb,
                    const float* __restrict__ bias,
                    float* __restrict__ C, long long sC, int ldc,
                    bf16* __restrict__ Cb, long long sCb, int ldcb,
                    int M, int N, int K, float alpha, int relu)
{
    __shared__ __align__(16) bf16 At[128 * LDA_LDS];
    __shared__ __align__(16) bf16 Bt[32 * LDB_LDS];

    const int z = blockIdx.z;
    A += z * sA;
    if (Bf) Bf += z * sB;
    if (Bh) Bh += z * sB;
    if (C)  C  += z * sC;
    if (Cb) Cb += z * sCb;

    const int m0   = blockIdx.y * 128;
    const int n0   = blockIdx.x * 128;
    const int tid  = threadIdx.x;
    const int lane = tid & 31;
    const int wave = tid >> 5;
    const int wm   = wave & 3;    // 32-row stripe
    const int wn   = wave >> 2;   // 64-col half

    // staging coordinates
    const int sar = tid >> 1;            // A row 0..127
    const int sac = (tid & 1) * 16;      // A col 0 or 16
    const int sbr = tid >> 3;            // B row 0..31
    const int sbc = (tid & 7) * 16;      // B col 0..112 step 16

    v8f acc[8] = {};

    for (int k0 = 0; k0 < K; k0 += 32) {
        // ---- prefetch next K tile (global_prefetch_b8) ----
        if (k0 + 32 < K) {
            if (m0 + sar < M)
                __builtin_prefetch(A + (long long)(m0 + sar) * lda + k0 + 32 + sac, 0, 3);
            if (Bf) __builtin_prefetch(Bf + (long long)(k0 + 32 + sbr) * ldb + n0 + sbc, 0, 3);
            else    __builtin_prefetch(Bh + (long long)(k0 + 32 + sbr) * ldb + n0 + sbc, 0, 3);
        }

        // ---- stage A tile (128x32 bf16): thread -> row, 16 cols ----
        {
            const int gm = m0 + sar;
            const int gk = k0 + sac;
            bf16* dst = &At[sar * LDA_LDS + sac];
            if (gm < M && gk + 16 <= K) {
                *reinterpret_cast<uint4*>(dst) =
                    *reinterpret_cast<const uint4*>(A + (long long)gm * lda + gk);
            } else {
                #pragma unroll
                for (int i = 0; i < 16; ++i)
                    dst[i] = (gm < M && gk + i < K)
                                 ? A[(long long)gm * lda + gk + i] : (bf16)0.f;
            }
        }
        // ---- stage B tile (32x128): thread -> row, 16 cols; cvt f32->bf16 ----
        {
            const int gk = k0 + sbr;
            const int gn = n0 + sbc;
            bf16* dst = &Bt[sbr * LDB_LDS + sbc];
            const bool full = (gk < K) && (gn + 16 <= N);
            if (Bf) {
                if (full) {
                    const float4* s4 =
                        reinterpret_cast<const float4*>(Bf + (long long)gk * ldb + gn);
                    bf16 tmp[16];
                    #pragma unroll
                    for (int q = 0; q < 4; ++q) {
                        const float4 f = s4[q];
                        tmp[q * 4 + 0] = (bf16)f.x;
                        tmp[q * 4 + 1] = (bf16)f.y;
                        tmp[q * 4 + 2] = (bf16)f.z;
                        tmp[q * 4 + 3] = (bf16)f.w;
                    }
                    reinterpret_cast<uint4*>(dst)[0] = reinterpret_cast<uint4*>(tmp)[0];
                    reinterpret_cast<uint4*>(dst)[1] = reinterpret_cast<uint4*>(tmp)[1];
                } else {
                    #pragma unroll
                    for (int i = 0; i < 16; ++i) {
                        float v = (gk < K && gn + i < N)
                                      ? Bf[(long long)gk * ldb + gn + i] : 0.f;
                        dst[i] = (bf16)v;
                    }
                }
            } else {
                if (full) {
                    const uint4* s4 =
                        reinterpret_cast<const uint4*>(Bh + (long long)gk * ldb + gn);
                    reinterpret_cast<uint4*>(dst)[0] = s4[0];
                    reinterpret_cast<uint4*>(dst)[1] = s4[1];
                } else {
                    #pragma unroll
                    for (int i = 0; i < 16; ++i)
                        dst[i] = (gk < K && gn + i < N)
                                     ? Bh[(long long)gk * ldb + gn + i] : (bf16)0.f;
                }
            }
        }
        __syncthreads();

        // ---- fragments (ISA 7.12.2 layouts) ----
        const int ar = lane & 15;
        const int ak = (lane >> 4) * 8;
        union Frag { v16bf v; uint4 q[2]; };
        Frag af0, af1;
        {
            const bf16* ab = &At[(wm * 32 + ar) * LDA_LDS + ak];
            af0.q[0] = *reinterpret_cast<const uint4*>(ab);
            af0.q[1] = *reinterpret_cast<const uint4*>(ab + 16);
            ab += 16 * LDA_LDS;
            af1.q[0] = *reinterpret_cast<const uint4*>(ab);
            af1.q[1] = *reinterpret_cast<const uint4*>(ab + 16);
        }
        #pragma unroll
        for (int nf = 0; nf < 4; ++nf) {
            const bf16* bb = &Bt[ak * LDB_LDS + wn * 64 + nf * 16 + ar];
            Frag bfr;
            #pragma unroll
            for (int s = 0; s < 16; ++s)
                bfr.v[s] = bb[((s & 7) + (s >> 3) * 16) * LDB_LDS];
            acc[nf] = __builtin_amdgcn_wmma_f32_16x16x32_bf16(
                false, af0.v, false, bfr.v, (short)0, acc[nf], false, false);
            acc[4 + nf] = __builtin_amdgcn_wmma_f32_16x16x32_bf16(
                false, af1.v, false, bfr.v, (short)0, acc[4 + nf], false, false);
        }
        __syncthreads();
    }

    // ---- epilogue: C/D layout: VGPR r -> M = r + 8*(lane>=16), N = lane&15 ----
    const int cl = lane & 15;
    #pragma unroll
    for (int mf = 0; mf < 2; ++mf) {
        const int rowBase = m0 + wm * 32 + mf * 16 + (lane >> 4) * 8;
        #pragma unroll
        for (int nf = 0; nf < 4; ++nf) {
            const int gn = n0 + wn * 64 + nf * 16 + cl;
            if (gn >= N) continue;
            const float bv = bias ? bias[gn] : 0.f;
            const v8f a = acc[mf * 4 + nf];
            #pragma unroll
            for (int r = 0; r < 8; ++r) {
                const int gm = rowBase + r;
                if (gm >= M) continue;
                float v = a[r] * alpha + bv;
                if (relu) v = v > 0.f ? v : 0.f;
                if (C)  C [(long long)gm * ldc  + gn] = v;
                if (Cb) Cb[(long long)gm * ldcb + gn] = (bf16)v;
            }
        }
    }
}

// ---------------------------------------------------------------------------
// x[b, 0..196, :] = image_embeds  (fp32 + bf16 mirror)
// ---------------------------------------------------------------------------
__global__ void copy_img(const float* __restrict__ img,
                         float* __restrict__ x, bf16* __restrict__ xb)
{
    const int i = blockIdx.x * 256 + threadIdx.x;
    const int total = 8 * 197 * 768;
    if (i >= total) return;
    const int d  = i % 768;
    const int rt = i / 768;            // b*197 + t
    const int b  = rt / 197;
    const int t  = rt % 197;
    const long long o = ((long long)(b * 274 + t)) * 768 + d;
    const float v = img[i];
    x[o]  = v;
    xb[o] = (bf16)v;
}

// ---------------------------------------------------------------------------
// cap_bf16[b*77+t, c] = tok_emb[ids[b,t], c] + pos_emb[t, c]
// ---------------------------------------------------------------------------
__global__ void embed_cap(const int* __restrict__ ids,
                          const float* __restrict__ tok,
                          const float* __restrict__ pos,
                          bf16* __restrict__ cap)
{
    const int i = blockIdx.x * 256 + threadIdx.x;
    const int total = 8 * 77 * 512;
    if (i >= total) return;
    const int c  = i % 512;
    const int bt = i / 512;
    const int t  = bt % 77;
    const int id = ids[bt];
    cap[i] = (bf16)(tok[(long long)id * 512 + c] + pos[t * 512 + c]);
}

// ---------------------------------------------------------------------------
// kt[b, d, s] = k[b, s, d]   (bf16, ldT = 288 so score-GEMM B rows are aligned)
// ---------------------------------------------------------------------------
__global__ void transpose_k(const bf16* __restrict__ k, bf16* __restrict__ kt)
{
    const int i = blockIdx.x * 256 + threadIdx.x;
    const int total = 8 * 274 * 768;
    if (i >= total) return;
    const int d  = i % 768;
    const int bs = i / 768;
    const int b  = bs / 274;
    const int s  = bs % 274;
    kt[((long long)b * 768 + d) * 288 + s] = k[i];
}

// ---------------------------------------------------------------------------
// Causal softmax over one score row (block = b*S + q). Writes bf16 attn row,
// zero-padded to ld so downstream WMMA K-tiles read zeros past q.
// ---------------------------------------------------------------------------
__global__ __launch_bounds__(256)
void softmax_causal(const float* __restrict__ scores, bf16* __restrict__ attn,
                    int S, int ld)
{
    const int row = blockIdx.x;
    const int q   = row % S;
    const float* srow = scores + (long long)row * ld;
    bf16* arow = attn + (long long)row * ld;
    const int tid = threadIdx.x;
    const int nvalid = q + 1;

    __shared__ float red[8];

    float mx = -3.0e38f;
    for (int i = tid; i < nvalid; i += 256) mx = fmaxf(mx, srow[i]);
    for (int o = 16; o > 0; o >>= 1) mx = fmaxf(mx, __shfl_xor(mx, o));
    if ((tid & 31) == 0) red[tid >> 5] = mx;
    __syncthreads();
    if (tid < 8) {
        float t = red[tid];
        for (int o = 4; o > 0; o >>= 1) t = fmaxf(t, __shfl_xor(t, o));
        red[tid] = t;
    }
    __syncthreads();
    mx = red[0];
    __syncthreads();

    float sum = 0.f;
    for (int i = tid; i < nvalid; i += 256) sum += __expf(srow[i] - mx);
    for (int o = 16; o > 0; o >>= 1) sum += __shfl_xor(sum, o);
    if ((tid & 31) == 0) red[tid >> 5] = sum;
    __syncthreads();
    if (tid < 8) {
        float t = red[tid];
        for (int o = 4; o > 0; o >>= 1) t += __shfl_xor(t, o);
        red[tid] = t;
    }
    __syncthreads();
    const float inv = 1.f / red[0];

    for (int i = tid; i < ld; i += 256) {
        const float v = (i < nvalid) ? __expf(srow[i] - mx) * inv : 0.f;
        arow[i] = (bf16)v;
    }
}

// ---------------------------------------------------------------------------
// x = LayerNorm(y + x) * g + b   (in place on x, bf16 mirror xb). D = 768.
// ---------------------------------------------------------------------------
__global__ __launch_bounds__(256)
void add_ln(const float* __restrict__ y, float* __restrict__ x,
            const float* __restrict__ g, const float* __restrict__ b,
            bf16* __restrict__ xb)
{
    const long long row = blockIdx.x;
    const float* yr = y + row * 768;
    float* xr = x + row * 768;
    const int tid = threadIdx.x;

    __shared__ float red[8];
    float v[3];
    float s = 0.f;
    #pragma unroll
    for (int j = 0; j < 3; ++j) {
        const int i = tid + j * 256;
        v[j] = yr[i] + xr[i];
        s += v[j];
    }
    for (int o = 16; o > 0; o >>= 1) s += __shfl_xor(s, o);
    if ((tid & 31) == 0) red[tid >> 5] = s;
    __syncthreads();
    if (tid < 8) {
        float t = red[tid];
        for (int o = 4; o > 0; o >>= 1) t += __shfl_xor(t, o);
        red[tid] = t;
    }
    __syncthreads();
    const float mu = red[0] * (1.f / 768.f);
    __syncthreads();

    float s2 = 0.f;
    #pragma unroll
    for (int j = 0; j < 3; ++j) { const float d = v[j] - mu; s2 += d * d; }
    for (int o = 16; o > 0; o >>= 1) s2 += __shfl_xor(s2, o);
    if ((tid & 31) == 0) red[tid >> 5] = s2;
    __syncthreads();
    if (tid < 8) {
        float t = red[tid];
        for (int o = 4; o > 0; o >>= 1) t += __shfl_xor(t, o);
        red[tid] = t;
    }
    __syncthreads();
    const float rstd = rsqrtf(red[0] * (1.f / 768.f) + 1e-5f);

    #pragma unroll
    for (int j = 0; j < 3; ++j) {
        const int i = tid + j * 256;
        const float o = (v[j] - mu) * rstd * g[i] + b[i];
        xr[i] = o;
        xb[row * 768 + i] = (bf16)o;
    }
}

// ---------------------------------------------------------------------------
extern "C" void kernel_launch(void* const* d_in, const int* in_sizes, int n_in,
                              void* d_out, int out_size, void* d_ws, size_t ws_size,
                              hipStream_t stream)
{
    (void)in_sizes; (void)n_in; (void)out_size; (void)ws_size;
    const float* img  = (const float*)d_in[0];
    const int*   ids  = (const int*)  d_in[1];
    const float* tok  = (const float*)d_in[2];
    const float* pos  = (const float*)d_in[3];
    const float* c2vw = (const float*)d_in[4];
    const float* c2vb = (const float*)d_in[5];
    const float* Wq = (const float*)d_in[6];   const float* bq = (const float*)d_in[7];
    const float* Wk = (const float*)d_in[8];   const float* bk = (const float*)d_in[9];
    const float* Wv = (const float*)d_in[10];  const float* bv = (const float*)d_in[11];
    const float* Wo = (const float*)d_in[12];  const float* bo = (const float*)d_in[13];
    const float* ln1g = (const float*)d_in[14]; const float* ln1b = (const float*)d_in[15];
    const float* W1 = (const float*)d_in[16];  const float* b1 = (const float*)d_in[17];
    const float* W2 = (const float*)d_in[18];  const float* b2 = (const float*)d_in[19];
    const float* ln2g = (const float*)d_in[20]; const float* ln2b = (const float*)d_in[21];
    const float* Ws = (const float*)d_in[22];  const float* bs = (const float*)d_in[23];
    float* out = (float*)d_out;

    const int Btc = 8, S = 274, D = 768, H = 2048, V = 49408, L = 6;
    const int MT = Btc * S;            // 2192 rows total
    const int LdS = 288;               // padded score/attn ld (multiple of 32)

    char* p = (char*)d_ws;
    auto carve = [&](size_t bytes) -> char* {
        char* r = p; p += (bytes + 255) & ~(size_t)255; return r;
    };
    float* x    = (float*)carve((size_t)MT * D * 4);
    bf16*  xb   = (bf16*) carve((size_t)MT * D * 2);
    bf16*  qb   = (bf16*) carve((size_t)MT * D * 2);
    bf16*  kb   = (bf16*) carve((size_t)MT * D * 2);
    bf16*  ktb  = (bf16*) carve((size_t)Btc * D * LdS * 2);
    bf16*  vb   = (bf16*) carve((size_t)MT * D * 2);
    float* sc   = (float*)carve((size_t)Btc * S * LdS * 4);
    bf16*  at   = (bf16*) carve((size_t)Btc * S * LdS * 2);
    bf16*  hidb = (bf16*) carve((size_t)MT * D * 2);
    float* ao   = (float*)carve((size_t)MT * D * 4);
    bf16*  hb   = (bf16*) carve((size_t)MT * H * 2);
    float* mo   = (float*)carve((size_t)MT * D * 4);
    bf16*  capb = (bf16*) carve((size_t)Btc * 77 * 512 * 2);

    auto gemm = [&](const bf16* A, long long sA, int lda,
                    const float* Bf, const bf16* Bh, long long sB, int ldb,
                    const float* bias, float* C, long long sC, int ldc,
                    bf16* Cb, long long sCb, int ldcb,
                    int M, int N, int K, float alpha, int relu, int Z) {
        dim3 grid((N + 127) / 128, (M + 127) / 128, Z);
        gemm_bf16_wmma<<<grid, 256, 0, stream>>>(A, sA, lda, Bf, Bh, sB, ldb,
                                                 bias, C, sC, ldc, Cb, sCb, ldcb,
                                                 M, N, K, alpha, relu);
    };

    // ---- embeddings ----
    copy_img<<<(Btc * 197 * D + 255) / 256, 256, 0, stream>>>(img, x, xb);
    embed_cap<<<(Btc * 77 * 512 + 255) / 256, 256, 0, stream>>>(ids, tok, pos, capb);
    // cap @ c2v_w + c2v_b -> x rows [197..273] per batch (batched over B)
    gemm(capb, (long long)77 * 512, 512, c2vw, nullptr, 0, D, c2vb,
         x + 197 * D, (long long)S * D, D, xb + 197 * D, (long long)S * D, D,
         77, D, 512, 1.f, 0, Btc);

    const float scale = 0.03608439182435161f;   // 1/sqrt(768)

    for (int i = 0; i < L; ++i) {
        const float* wq = Wq + (long long)i * D * D;
        const float* wk = Wk + (long long)i * D * D;
        const float* wv = Wv + (long long)i * D * D;
        const float* wo = Wo + (long long)i * D * D;
        const float* w1 = W1 + (long long)i * D * H;
        const float* w2 = W2 + (long long)i * H * D;

        // Q, K, V projections (bf16 outputs feed attention GEMMs)
        gemm(xb, 0, D, wq, nullptr, 0, D, bq + i * D,
             nullptr, 0, 0, qb, 0, D, MT, D, D, 1.f, 0, 1);
        gemm(xb, 0, D, wk, nullptr, 0, D, bk + i * D,
             nullptr, 0, 0, kb, 0, D, MT, D, D, 1.f, 0, 1);
        gemm(xb, 0, D, wv, nullptr, 0, D, bv + i * D,
             nullptr, 0, 0, vb, 0, D, MT, D, D, 1.f, 0, 1);

        // K -> K^T  so scores become a row-major GEMM
        transpose_k<<<(MT * D + 255) / 256, 256, 0, stream>>>(kb, ktb);

        // scores = scale * Q x K^T   [274 x 274] per batch
        gemm(qb, (long long)S * D, D, nullptr, ktb, (long long)D * LdS, LdS,
             nullptr, sc, (long long)S * LdS, LdS, nullptr, 0, 0,
             S, S, D, scale, 0, Btc);

        softmax_causal<<<MT, 256, 0, stream>>>(sc, at, S, LdS);

        // hid = attn x V   (K = 274, attn rows zero-padded to 288)
        gemm(at, (long long)S * LdS, LdS, nullptr, vb, (long long)S * D, D,
             nullptr, nullptr, 0, 0, hidb, (long long)S * D, D,
             S, D, S, 1.f, 0, Btc);

        // attn output projection + residual LN
        gemm(hidb, 0, D, wo, nullptr, 0, D, bo + i * D,
             ao, 0, D, nullptr, 0, 0, MT, D, D, 1.f, 0, 1);
        add_ln<<<MT, 256, 0, stream>>>(ao, x, ln1g + i * D, ln1b + i * D, xb);

        // MLP
        gemm(xb, 0, D, w1, nullptr, 0, H, b1 + i * H,
             nullptr, 0, 0, hb, 0, H, MT, H, D, 1.f, 1, 1);
        gemm(hb, 0, H, w2, nullptr, 0, D, b2 + i * D,
             mo, 0, D, nullptr, 0, 0, MT, D, H, 1.f, 0, 1);
        add_ln<<<MT, 256, 0, stream>>>(mo, x, ln2g + i * D, ln2b + i * D, xb);
    }

    // vocab projection straight to d_out
    gemm(xb, 0, D, Ws, nullptr, 0, V, bs,
         out, 0, V, nullptr, 0, 0, MT, V, D, 1.f, 0, 1);
}
